// AverageAttention_72748156060160
// MI455X (gfx1250) — compile-verified, matrix operand order checked by reading, not run on hardware
//
#include <hip/hip_runtime.h>
#include <hip/hip_bf16.h>

// Problem constants (match reference)
constexpr int B  = 8;
constexpr int L  = 2048;
constexpr int D  = 1024;
constexpr int K2 = 2 * D;         // GEMM K dimension = 2048
constexpr int M  = B * L;         // GEMM M dimension = 16384
constexpr int NCHUNK = 8;         // scan chunks along L
constexpr int CLEN   = L / NCHUNK;

typedef __attribute__((ext_vector_type(16))) __bf16 v16bf;
typedef __attribute__((ext_vector_type(8)))  float  v8f;

// ---------- helpers ----------

__device__ __forceinline__ unsigned short f32_to_bf16_rne(float f) {
    unsigned u = __float_as_uint(f);
    unsigned r = (u + 0x7FFFu + ((u >> 16) & 1u)) >> 16;
    return (unsigned short)r;
}

union frag_u { v16bf v; uint4 q[2]; };

// Load one 16x32 bf16 WMMA fragment (A layout; B uses the same pattern with
// lane = column since W is stored row-major (out, in) and B = W^T).
// Lanes 0-15: rows 0-15, K = 0..7 (VGPR0-3) and 16..23 (VGPR4-7).
// Lanes 16-31: same rows, K offset +8.
__device__ __forceinline__ v16bf load_frag(const unsigned short* __restrict__ base,
                                           int row, int stride, int kb, int lane) {
    const int klo = (lane & 16) ? 8 : 0;
    const unsigned short* p = base + (size_t)row * stride + kb + klo;
    frag_u u;
    u.q[0] = *(const uint4*)(p);        // K = klo .. klo+7
    u.q[1] = *(const uint4*)(p + 16);   // K = 16+klo .. 16+klo+7
    return u.v;
}

// Fast sigmoid: v_mul + v_exp + v_add + v_rcp (no IEEE div expansion).
__device__ __forceinline__ float sigmoidf(float g) {
    return __builtin_amdgcn_rcpf(1.0f + __expf(-g));
}

// ---------- kernel 1: per-chunk sums for the split scan ----------
__global__ void chunk_sum_kernel(const float* __restrict__ x, float* __restrict__ partial) {
    int tid = blockIdx.x * blockDim.x + threadIdx.x;   // B*NCHUNK*D threads
    int d = tid % D;
    int c = (tid / D) % NCHUNK;
    int b = tid / (D * NCHUNK);
    const float* p = x + ((size_t)(b * L + c * CLEN)) * D + d;
    float s = 0.0f;
    for (int l = 0; l < CLEN; ++l) s += p[(size_t)l * D];
    partial[tid] = s;
}

// ---------- kernel 2: in-chunk scan -> avg (f32 out) + bf16 cat buffer ----------
__global__ void scan_kernel(const float* __restrict__ x, const float* __restrict__ partial,
                            float* __restrict__ avg, unsigned short* __restrict__ catb) {
    int tid = blockIdx.x * blockDim.x + threadIdx.x;
    int d = tid % D;
    int c = (tid / D) % NCHUNK;
    int b = tid / (D * NCHUNK);

    float run = 0.0f;
    for (int c2 = 0; c2 < c; ++c2)
        run += partial[(size_t)(b * NCHUNK + c2) * D + d];

    size_t base = ((size_t)(b * L + c * CLEN)) * D + d;
    for (int l = 0; l < CLEN; ++l) {
        size_t gi = base + (size_t)l * D;
        float xv = x[gi];
        run += xv;
        int t = c * CLEN + l;
        float av = run / (float)(t + 1);
        avg[gi] = av;
        size_t m = (size_t)(b * L + t);
        catb[m * K2 + d]     = f32_to_bf16_rne(xv);
        catb[m * K2 + D + d] = f32_to_bf16_rne(av);
    }
}

// ---------- kernel 3: W fp32 -> bf16 ----------
__global__ void convert_w_kernel(const float* __restrict__ W, unsigned short* __restrict__ Wb) {
    int i = (blockIdx.x * blockDim.x + threadIdx.x) * 4;
    float4 f = *(const float4*)(W + i);
    ushort4 o;
    o.x = f32_to_bf16_rne(f.x);
    o.y = f32_to_bf16_rne(f.y);
    o.z = f32_to_bf16_rne(f.z);
    o.w = f32_to_bf16_rne(f.w);
    *(ushort4*)(Wb + i) = o;
}

// ---------- kernel 4: WMMA GEMM with fused sigmoid gating epilogue ----------

__device__ __forceinline__ void loadB(v16bf bf[2][2], const unsigned short* __restrict__ Wb,
                                      int nbase, int kb, int lane) {
#pragma unroll
    for (int np = 0; np < 2; ++np) {
        const int col = nbase + np * 16 + (lane & 15);
        bf[np][0] = load_frag(Wb, col,     K2, kb, lane);   // input-gate rows
        bf[np][1] = load_frag(Wb, col + D, K2, kb, lane);   // forget-gate rows
    }
}

__device__ __forceinline__ void wmma_step(v8f acc[4][2][2], const v16bf a[4],
                                          const v16bf bf[2][2]) {
#pragma unroll
    for (int i = 0; i < 4; ++i)
#pragma unroll
        for (int np = 0; np < 2; ++np) {
            acc[i][np][0] = __builtin_amdgcn_wmma_f32_16x16x32_bf16(
                false, a[i], false, bf[np][0], (short)0, acc[i][np][0], false, false);
            acc[i][np][1] = __builtin_amdgcn_wmma_f32_16x16x32_bf16(
                false, a[i], false, bf[np][1], (short)0, acc[i][np][1], false, false);
        }
}

// Block: 256 threads = 8 waves (2 M-waves x 4 N-waves). Block tile: M=128,
// 128 gate columns. Each wave: 4 M-subtiles x 2 column-pairs x 2 gate halves
// -> 16 accumulators; 16 v_wmma_f32_16x16x32_bf16 per K=32 step. K loop is
// unrolled x2 with ping-pong A fragments so loads overlap WMMA execution.
__global__ __launch_bounds__(256)
void gemm_gate_kernel(const unsigned short* __restrict__ catb,
                      const unsigned short* __restrict__ Wb,
                      const float* __restrict__ bias,
                      const float* __restrict__ x,
                      const float* __restrict__ avg,
                      float* __restrict__ out) {
    const int lane = threadIdx.x & 31;
    const int wave = threadIdx.x >> 5;
    const int wm = wave & 1;        // 0..1  -> 64-row strip
    const int wn = wave >> 1;       // 0..3  -> 32-col strip (2 pair-tiles)
    const int mbase = blockIdx.x * 128 + wm * 64;
    const int nbase = blockIdx.y * 128 + wn * 32;  // column in [0, D)

    // acc[msub][npair][half], initialized with bias (depends only on column)
    v8f acc[4][2][2];
#pragma unroll
    for (int np = 0; np < 2; ++np) {
        const int col = nbase + np * 16 + (lane & 15);
        const float bi = bias[col];
        const float bf_ = bias[col + D];
#pragma unroll
        for (int i = 0; i < 4; ++i)
#pragma unroll
            for (int r = 0; r < 8; ++r) { acc[i][np][0][r] = bi; acc[i][np][1][r] = bf_; }
    }

    const int arow = mbase + (lane & 15);

    v16bf a0[4], a1[4];
#pragma unroll
    for (int i = 0; i < 4; ++i) a0[i] = load_frag(catb, arow + i * 16, K2, 0, lane);

    for (int kb = 0; kb < K2; kb += 64) {
        v16bf bf[2][2];
        loadB(bf, Wb, nbase, kb, lane);
#pragma unroll
        for (int i = 0; i < 4; ++i) a1[i] = load_frag(catb, arow + i * 16, K2, kb + 32, lane);
        wmma_step(acc, a0, bf);

        loadB(bf, Wb, nbase, kb + 32, lane);
        if (kb + 64 < K2) {
#pragma unroll
            for (int i = 0; i < 4; ++i) a0[i] = load_frag(catb, arow + i * 16, K2, kb + 64, lane);
        }
        wmma_step(acc, a1, bf);
    }

    // Fused gating epilogue. C/D layout: VGPR r, lanes 0-15 -> row r, lanes
    // 16-31 -> row r+8; column = lane & 15.
    const int rowoff = (lane & 16) ? 8 : 0;
#pragma unroll
    for (int i = 0; i < 4; ++i) {
#pragma unroll
        for (int np = 0; np < 2; ++np) {
            const int col = nbase + np * 16 + (lane & 15);
#pragma unroll
            for (int r = 0; r < 8; ++r) {
                int m = mbase + i * 16 + rowoff + r;
                size_t idx = (size_t)m * D + col;
                float xv = x[idx];
                float av = avg[idx];
                out[idx] = sigmoidf(acc[i][np][0][r]) * xv + sigmoidf(acc[i][np][1][r]) * av;
            }
        }
    }
}

// ---------- launcher ----------
extern "C" void kernel_launch(void* const* d_in, const int* in_sizes, int n_in,
                              void* d_out, int out_size, void* d_ws, size_t ws_size,
                              hipStream_t stream) {
    const float* x    = (const float*)d_in[0];  // (B, L, D)
    const float* W    = (const float*)d_in[1];  // (2D, 2D)
    const float* bias = (const float*)d_in[2];  // (2D,)

    float* gate_out = (float*)d_out;                     // first output
    float* avg_out  = gate_out + (size_t)M * D;          // second output

    char* ws = (char*)d_ws;
    unsigned short* catb = (unsigned short*)ws;                          // 64 MB
    unsigned short* Wb   = (unsigned short*)(ws + (size_t)M * K2 * 2);   //  8 MB
    float* partial = (float*)(ws + (size_t)M * K2 * 2 + (size_t)K2 * K2 * 2); // 256 KB

    int scan_threads = B * NCHUNK * D;   // 65536
    chunk_sum_kernel<<<scan_threads / 256, 256, 0, stream>>>(x, partial);
    scan_kernel<<<scan_threads / 256, 256, 0, stream>>>(x, partial, avg_out, catb);
    convert_w_kernel<<<(K2 * K2 / 4) / 256, 256, 0, stream>>>(W, Wb);

    dim3 grid(M / 128, D / 128);         // 128 x 8
    gemm_gate_kernel<<<grid, 256, 0, stream>>>(catb, Wb, bias, x, avg_out, gate_out);
}
